// Transformer_GINEConv_NN_9603546874328
// MI455X (gfx1250) — matrix operand order repaired and checked by reading, not compiled, not run on hardware
//
#include <hip/hip_runtime.h>
#include <hip/hip_bf16.h>

typedef __attribute__((ext_vector_type(16))) _Float16 v16h;
typedef __attribute__((ext_vector_type(8)))  _Float16 v8h;
typedef __attribute__((ext_vector_type(8)))  float    v8f;
typedef __attribute__((ext_vector_type(4)))  float    v4f;

#define NN   8192
#define EE   524288
#define HH   128
#define NCC  16

// ---------------------------------------------------------------------------
// zero workspace (agg) -- harness does not re-zero between replays
// ---------------------------------------------------------------------------
__global__ __launch_bounds__(256) void k_zero(float* __restrict__ p) {
    int i = (blockIdx.x * 256 + threadIdx.x) * 4;
    *(v4f*)(p + i) = (v4f){0.f, 0.f, 0.f, 0.f};
}

// ---------------------------------------------------------------------------
// transpose+convert ee_W2 (128x128 f32 row-major, [k][n]) -> Wt f16 [n][k]
// so WMMA B-fragments become contiguous 32B runs per lane.
// ---------------------------------------------------------------------------
__global__ __launch_bounds__(256) void k_w2t(const float* __restrict__ W,
                                             _Float16* __restrict__ Wt) {
    int i = blockIdx.x * 256 + threadIdx.x;   // 0..16383
    int n = i >> 7, k = i & 127;
    Wt[n * 128 + k] = (_Float16)W[k * 128 + n];
}

// ---------------------------------------------------------------------------
// node encoder: coded_x = Lin(32->128) -> PReLU -> Lin(128->128)
// block = 128 threads handles 8 nodes
// ---------------------------------------------------------------------------
__global__ __launch_bounds__(128) void k_node_enc(
    const float* __restrict__ x,
    const float* __restrict__ W1, const float* __restrict__ b1,
    const float* __restrict__ aP,
    const float* __restrict__ W2, const float* __restrict__ b2,
    float* __restrict__ coded_x) {
    __shared__ float sx[8][32];
    __shared__ float sh[8][128];
    const int t  = threadIdx.x;
    const int nb = blockIdx.x * 8;
    for (int idx = t; idx < 256; idx += 128) ((float*)sx)[idx] = x[nb * 32 + idx];
    __syncthreads();
    float acc[8];
    #pragma unroll
    for (int nl = 0; nl < 8; ++nl) acc[nl] = b1[t];
    for (int k = 0; k < 32; ++k) {
        float w = W1[k * 128 + t];
        #pragma unroll
        for (int nl = 0; nl < 8; ++nl) acc[nl] += sx[nl][k] * w;
    }
    const float a = aP[0];
    #pragma unroll
    for (int nl = 0; nl < 8; ++nl) {
        float h = acc[nl];
        sh[nl][t] = (h >= 0.f) ? h : a * h;
    }
    __syncthreads();
    #pragma unroll
    for (int nl = 0; nl < 8; ++nl) acc[nl] = b2[t];
    for (int k = 0; k < 128; ++k) {
        float w = W2[k * 128 + t];
        #pragma unroll
        for (int nl = 0; nl < 8; ++nl) acc[nl] += sh[nl][k] * w;
    }
    #pragma unroll
    for (int nl = 0; nl < 8; ++nl) coded_x[(nb + nl) * 128 + t] = acc[nl];
}

// ---------------------------------------------------------------------------
// fused edge pipeline: edgeEncoder layer1 (4->128, PReLU) to LDS f16,
// layer2 (128->128) via v_wmma_f32_16x16x32_f16, then
// msg = relu(coded_x[src] + e), agg[dst] += msg  (global f32 atomics).
// block = 256 threads (8 wave32) handles 128 edges; each wave owns 16 edges.
// ---------------------------------------------------------------------------
__global__ __launch_bounds__(256) void k_edge(
    const float* __restrict__ edge_attr, const int* __restrict__ ei,
    const float* __restrict__ W1, const float* __restrict__ b1,
    const float* __restrict__ aP,
    const _Float16* __restrict__ W2t, const float* __restrict__ b2,
    const float* __restrict__ coded_x, float* __restrict__ agg) {
    constexpr int LSTR = 136;                       // pad: 272B row -> no bank camp
    __shared__ __align__(16) _Float16 h1[128 * LSTR];
    __shared__ float sea[512];
    const int  t     = threadIdx.x;
    const long ebase = (long)blockIdx.x * 128;

    sea[t]       = edge_attr[ebase * 4 + t];
    sea[t + 256] = edge_attr[ebase * 4 + 256 + t];
    __syncthreads();

    // layer1 + PReLU, f16 into LDS
    const int   f  = t & 127;
    const float w0 = W1[f], w1 = W1[128 + f], w2 = W1[256 + f], w3 = W1[384 + f];
    const float bb = b1[f], a = aP[0];
    for (int el = (t >> 7); el < 128; el += 2) {
        float h = bb + sea[el * 4] * w0 + sea[el * 4 + 1] * w1 +
                       sea[el * 4 + 2] * w2 + sea[el * 4 + 3] * w3;
        h = (h >= 0.f) ? h : a * h;
        h1[el * LSTR + f] = (_Float16)h;
    }
    __syncthreads();

    const int lane = t & 31, wave = t >> 5;
    const int lo = lane & 15, hi = lane >> 4;
    const int erow = wave * 16;
    const long eg0 = ebase + erow + 8 * hi;
    int srcs[8], dsts[8];
    #pragma unroll
    for (int r = 0; r < 8; ++r) {
        long eg = eg0 + r;
        srcs[r] = ei[eg];
        dsts[r] = ei[(long)EE + eg];
    }

    // A fragments (16x32 f16 per K-chunk) from LDS
    v16h A[4];
    #pragma unroll
    for (int kc = 0; kc < 4; ++kc) {
        const _Float16* pa = h1 + (erow + lo) * LSTR + kc * 32 + hi * 16;
        v8h x0 = *(const v8h*)pa;
        v8h x1 = *(const v8h*)(pa + 8);
        A[kc] = __builtin_shufflevector(x0, x1, 0,1,2,3,4,5,6,7,8,9,10,11,12,13,14,15);
    }

    #pragma unroll
    for (int jt = 0; jt < 8; ++jt) {
        const int   n    = jt * 16 + lo;
        const float bias = b2[n];
        v8f acc;
        #pragma unroll
        for (int i = 0; i < 8; ++i) acc[i] = bias;
        #pragma unroll
        for (int kc = 0; kc < 4; ++kc) {
            v16h B = *(const v16h*)(W2t + n * 128 + kc * 32 + hi * 16);
            acc = __builtin_amdgcn_wmma_f32_16x16x32_f16(
                false, A[kc], false, B, (short)0, acc, false, false);
        }
        #pragma unroll
        for (int r = 0; r < 8; ++r) {
            float msg = acc[r] + coded_x[(long)srcs[r] * 128 + n];
            msg = fmaxf(msg, 0.f);
            atomicAdd(&agg[(long)dsts[r] * 128 + n], msg);
        }
    }
}

// ---------------------------------------------------------------------------
// GINE MLP: z = Lin(PReLU(Lin((1+eps)*coded_x + agg))); also emit z in f16
// ---------------------------------------------------------------------------
__global__ __launch_bounds__(128) void k_gmlp(
    const float* __restrict__ coded_x, const float* __restrict__ agg,
    const float* __restrict__ epsP,
    const float* __restrict__ W1, const float* __restrict__ b1,
    const float* __restrict__ aP,
    const float* __restrict__ W2, const float* __restrict__ b2,
    float* __restrict__ z, _Float16* __restrict__ zh) {
    __shared__ float si[8][128];
    __shared__ float sh[8][128];
    const int   t  = threadIdx.x;
    const int   nb = blockIdx.x * 8;
    const float oe = 1.0f + epsP[0];
    #pragma unroll
    for (int nl = 0; nl < 8; ++nl)
        si[nl][t] = oe * coded_x[(nb + nl) * 128 + t] + agg[(nb + nl) * 128 + t];
    __syncthreads();
    float acc[8];
    #pragma unroll
    for (int nl = 0; nl < 8; ++nl) acc[nl] = b1[t];
    for (int k = 0; k < 128; ++k) {
        float w = W1[k * 128 + t];
        #pragma unroll
        for (int nl = 0; nl < 8; ++nl) acc[nl] += si[nl][k] * w;
    }
    const float a = aP[0];
    #pragma unroll
    for (int nl = 0; nl < 8; ++nl) {
        float h = acc[nl];
        sh[nl][t] = (h >= 0.f) ? h : a * h;
    }
    __syncthreads();
    #pragma unroll
    for (int nl = 0; nl < 8; ++nl) acc[nl] = b2[t];
    for (int k = 0; k < 128; ++k) {
        float w = W2[k * 128 + t];
        #pragma unroll
        for (int nl = 0; nl < 8; ++nl) acc[nl] += sh[nl][k] * w;
    }
    #pragma unroll
    for (int nl = 0; nl < 8; ++nl) {
        z[(nb + nl) * 128 + t]  = acc[nl];
        zh[(nb + nl) * 128 + t] = (_Float16)acc[nl];
    }
}

// ---------------------------------------------------------------------------
// A_hat = z @ z^T  (f16 WMMA, f32 acc). z_h is 2MB -> L2-resident; kernel is
// bound by the 256MB output stream. Each wave computes a 16x64 strip.
// ---------------------------------------------------------------------------
__global__ __launch_bounds__(256) void k_aat(const _Float16* __restrict__ zh,
                                             float* __restrict__ out) {
    const int t = threadIdx.x, lane = t & 31, wave = t >> 5;
    const int lo = lane & 15, hi = lane >> 4;
    const int it  = blockIdx.y * 8 + wave;   // row tile 0..511
    const int jt0 = blockIdx.x * 4;          // col tile base

    const _Float16* arow = zh + (it * 16 + lo) * 128;
    v16h A[4];
    #pragma unroll
    for (int kc = 0; kc < 4; ++kc) {
        const _Float16* pa = arow + kc * 32 + hi * 16;
        v8h x0 = *(const v8h*)pa;
        v8h x1 = *(const v8h*)(pa + 8);
        A[kc] = __builtin_shufflevector(x0, x1, 0,1,2,3,4,5,6,7,8,9,10,11,12,13,14,15);
    }
    #pragma unroll
    for (int jj = 0; jj < 4; ++jj) {
        const int jt = jt0 + jj;
        v8f acc;
        #pragma unroll
        for (int i = 0; i < 8; ++i) acc[i] = 0.f;
        #pragma unroll
        for (int kc = 0; kc < 4; ++kc) {
            // B row n of z^T is just row (jt*16+n) of z: contiguous 32B run
            v16h B = *(const v16h*)(zh + (jt * 16 + lo) * 128 + kc * 32 + hi * 16);
            acc = __builtin_amdgcn_wmma_f32_16x16x32_f16(
                false, A[kc], false, B, (short)0, acc, false, false);
        }
        float* o = out + (long)(it * 16 + 8 * hi) * NN + jt * 16 + lo;
        #pragma unroll
        for (int r = 0; r < 8; ++r) o[(long)r * NN] = acc[r];
    }
}

// ---------------------------------------------------------------------------
// p = softmax(z @ p_W + p_b): one thread per node
// ---------------------------------------------------------------------------
__global__ __launch_bounds__(128) void k_p(const float* __restrict__ z,
                                           const float* __restrict__ pW,
                                           const float* __restrict__ pb,
                                           float* __restrict__ p) {
    __shared__ float sW[128 * 16];
    __shared__ float sb[16];
    const int t = threadIdx.x;
    for (int i = t; i < 2048; i += 128) sW[i] = pW[i];
    if (t < 16) sb[t] = pb[t];
    __syncthreads();
    const int node = blockIdx.x * 128 + t;
    const float* zr = z + (long)node * 128;
    float lg[16];
    #pragma unroll
    for (int c = 0; c < 16; ++c) lg[c] = sb[c];
    for (int k = 0; k < 128; ++k) {
        float zv = zr[k];
        #pragma unroll
        for (int c = 0; c < 16; ++c) lg[c] += zv * sW[k * 16 + c];
    }
    float m = lg[0];
    #pragma unroll
    for (int c = 1; c < 16; ++c) m = fmaxf(m, lg[c]);
    float s = 0.f;
    #pragma unroll
    for (int c = 0; c < 16; ++c) { lg[c] = __expf(lg[c] - m); s += lg[c]; }
    const float inv = 1.f / s;
    #pragma unroll
    for (int c = 0; c < 16; ++c) p[(long)node * 16 + c] = lg[c] * inv;
}

// ---------------------------------------------------------------------------
extern "C" void kernel_launch(void* const* d_in, const int* in_sizes, int n_in,
                              void* d_out, int out_size, void* d_ws, size_t ws_size,
                              hipStream_t stream) {
    const float* x    = (const float*)d_in[0];
    const float* ea   = (const float*)d_in[1];
    const int*   ei   = (const int*)d_in[2];
    const float* neW1 = (const float*)d_in[3];
    const float* neb1 = (const float*)d_in[4];
    const float* nea  = (const float*)d_in[5];
    const float* neW2 = (const float*)d_in[6];
    const float* neb2 = (const float*)d_in[7];
    const float* eeW1 = (const float*)d_in[8];
    const float* eeb1 = (const float*)d_in[9];
    const float* eea  = (const float*)d_in[10];
    const float* eeW2 = (const float*)d_in[11];
    const float* eeb2 = (const float*)d_in[12];
    const float* gW1  = (const float*)d_in[13];
    const float* gb1  = (const float*)d_in[14];
    const float* ga   = (const float*)d_in[15];
    const float* gW2  = (const float*)d_in[16];
    const float* gb2  = (const float*)d_in[17];
    const float* eps  = (const float*)d_in[18];
    const float* pW   = (const float*)d_in[19];
    const float* pb   = (const float*)d_in[20];
    (void)in_sizes; (void)n_in; (void)out_size; (void)ws_size;

    float* out   = (float*)d_out;
    float* A_hat = out;                               // 8192*8192
    float* pout  = out + (long)NN * NN;               // 8192*16
    float* zout  = pout + (long)NN * NCC;             // 8192*128

    char* ws = (char*)d_ws;
    float*    coded_x = (float*)ws;                              // 4 MB
    float*    agg     = (float*)(ws + (size_t)4  * 1024 * 1024); // 4 MB
    _Float16* zh      = (_Float16*)(ws + (size_t)8 * 1024 * 1024);  // 2 MB
    _Float16* W2t     = (_Float16*)(ws + (size_t)10 * 1024 * 1024); // 32 KB

    k_zero<<<1024, 256, 0, stream>>>(agg);                      // NN*HH floats
    k_node_enc<<<NN / 8, 128, 0, stream>>>(x, neW1, neb1, nea, neW2, neb2, coded_x);
    k_w2t<<<64, 256, 0, stream>>>(eeW2, W2t);
    k_edge<<<EE / 128, 256, 0, stream>>>(ea, ei, eeW1, eeb1, eea, W2t, eeb2,
                                         coded_x, agg);
    k_gmlp<<<NN / 8, 128, 0, stream>>>(coded_x, agg, eps, gW1, gb1, ga, gW2, gb2,
                                       zout, zh);
    k_aat<<<dim3(128, 64), 256, 0, stream>>>(zh, A_hat);
    k_p<<<NN / 128, 128, 0, stream>>>(zout, pW, pb, pout);
}
